// GAT_block_36773509988937
// MI455X (gfx1250) — matrix operand order repaired and checked by reading, not compiled, not run on hardware
//
#include <hip/hip_runtime.h>
#include <stdint.h>

// ---------------------------------------------------------------------------
// GATv2 block for MI455X (gfx1250, wave32).
// Dense GEMMs use v_wmma_f32_16x16x32_bf16 (fp32 accumulate) with 16x64
// register blocking (A fragment reused across 4 N-tiles). Edge phase uses
// L2-resident gathers + native f32 global atomics. conv_bias cancels in BN1.
// ---------------------------------------------------------------------------

#define N_NODES 50000
#define DIM     256
#define N_EDGES 800000
#define DHID    512
#define EPSBN   1e-5f

typedef __bf16 bf16_t;
typedef __bf16 bf16x16 __attribute__((ext_vector_type(16)));
typedef __bf16 bf16x8  __attribute__((ext_vector_type(8)));
typedef float  f32x8   __attribute__((ext_vector_type(8)));

__device__ __forceinline__ bf16_t f2bf(float f) {
  unsigned u = __float_as_uint(f);
  u += 0x7FFFu + ((u >> 16) & 1u);          // round-to-nearest-even
  unsigned short h = (unsigned short)(u >> 16);
  return __builtin_bit_cast(bf16_t, h);
}

// order-preserving float <-> uint encoding (for atomicMax on floats)
__device__ __forceinline__ unsigned f2ord(float x) {
  unsigned u = __float_as_uint(x);
  return (u & 0x80000000u) ? ~u : (u | 0x80000000u);
}
__device__ __forceinline__ float ord2f(unsigned v) {
  unsigned u = (v & 0x80000000u) ? (v & 0x7FFFFFFFu) : ~v;
  return __uint_as_float(u);
}

// ---------------------------------------------------------------- conversions
__global__ void k_f32_to_bf16(const float* __restrict__ src,
                              bf16_t* __restrict__ dst, long n) {
  long i = (long)blockIdx.x * blockDim.x + threadIdx.x;
  long stride = (long)gridDim.x * blockDim.x;
  for (; i < n; i += stride) dst[i] = f2bf(src[i]);
}

// w[K][Nout] (f32, row-major)  ->  wt[Nout][K] (bf16)   (B-operand layout)
__global__ void k_transpose_bf16(const float* __restrict__ w,
                                 bf16_t* __restrict__ wt, int K, int Nout) {
  int i = blockIdx.x * blockDim.x + threadIdx.x;
  if (i >= K * Nout) return;
  int n = i / K, k = i - n * K;
  wt[(size_t)n * K + k] = f2bf(w[(size_t)k * Nout + n]);
}

// ---------------------------------------------------------------- WMMA GEMM
// C[M,Nout] = A[M,K](bf16) * Bt[Nout,K](bf16)^T, fp32 accumulate.
// Each wave computes a 16(M) x 64(N) strip = 4 tiles, reusing the A fragment.
// mode 0: + (n<256 ? bias0[n] : bias1[n-256])                     (GEMM1)
// mode 1: + bias0[n], LeakyReLU(0.01)                             (GEMM2)
// mode 2: + bias0[n], LeakyReLU(0.01), + resid[row*256+n]         (GEMM3)
__global__ void k_wmma_gemm(const bf16_t* __restrict__ A,
                            const bf16_t* __restrict__ Bt,
                            float* __restrict__ C,
                            int M, int Nout, int K, int mode,
                            const float* __restrict__ bias0,
                            const float* __restrict__ bias1,
                            const float* __restrict__ resid) {
  const int lane  = threadIdx.x & 31;
  const int wave  = threadIdx.x >> 5;
  const int strips = Nout >> 6;                  // N strips of 64
  int strip = blockIdx.x * (blockDim.x >> 5) + wave;
  if (strip >= (M >> 4) * strips) return;        // wave-uniform exit
  int tm = strip / strips;
  int tn = strip - tm * strips;
  const int mBase = tm << 4, nBase = tn << 6;
  const int khalf = lane >> 4;                   // which K-half this lane owns
  const int mRow  = mBase + (lane & 15);
  const bf16_t* aRow = A  + (size_t)mRow * K + (khalf << 4);            // +khalf*16
  const bf16_t* bRow = Bt + (size_t)(nBase + (lane & 15)) * K + (khalf << 4); // +khalf*16
  const size_t bTile = (size_t)16 * K;           // next 16 output columns

  f32x8 acc[4];
#pragma unroll
  for (int j = 0; j < 4; ++j) acc[j] = (f32x8){0.f,0.f,0.f,0.f,0.f,0.f,0.f,0.f};

  for (int k0 = 0; k0 < K; k0 += 32) {
    bf16x8 alo = *(const bf16x8*)(aRow + k0);        // K: 0..7  (or 16..23)
    bf16x8 ahi = *(const bf16x8*)(aRow + k0 + 8);    // K: 8..15 (or 24..31)
    bf16x16 av = __builtin_shufflevector(alo, ahi,
        0,1,2,3,4,5,6,7,8,9,10,11,12,13,14,15);
    bf16x16 bv[4];
#pragma unroll
    for (int j = 0; j < 4; ++j)
      bv[j] = *(const bf16x16*)(bRow + (size_t)j * bTile + k0);
#pragma unroll
    for (int j = 0; j < 4; ++j)
      acc[j] = __builtin_amdgcn_wmma_f32_16x16x32_bf16(
          false, av, false, bv[j], (short)0, acc[j], false, false);
  }

#pragma unroll
  for (int j = 0; j < 4; ++j) {
    const int n = nBase + (j << 4) + (lane & 15);
    float bias = (mode == 0) ? ((n < 256) ? bias0[n] : bias1[n - 256]) : bias0[n];
#pragma unroll
    for (int r = 0; r < 8; ++r) {
      int row = mBase + r + (khalf << 3);
      float v = acc[j][r] + bias;
      if (mode >= 1) v = v > 0.f ? v : 0.01f * v;
      if (mode == 2) v += resid[(size_t)row * DIM + n];
      C[(size_t)row * Nout + n] = v;
    }
  }
}

// -------------------------------------------------------------- edge kernels
// One wave per edge; lane handles 8 channels; head = lane>>2.
__global__ void k_edge_logits(const int* __restrict__ ei,
                              const float* __restrict__ ea,
                              const float* __restrict__ xlr,
                              const float* __restrict__ We,   // [16][256]
                              const float* __restrict__ att,  // [256]
                              float* __restrict__ logits,
                              unsigned* __restrict__ segmax) {
  __shared__ float sWe[16 * 256];
  __shared__ float sAtt[256];
  for (int i = threadIdx.x; i < 16 * 256; i += blockDim.x) sWe[i] = We[i];
  if (threadIdx.x < 256) sAtt[threadIdx.x] = att[threadIdx.x];
  __syncthreads();
  int e = blockIdx.x * (blockDim.x >> 5) + (threadIdx.x >> 5);
  if (e >= N_EDGES) return;
  int lane = threadIdx.x & 31;
  int src = ei[e], dst = ei[N_EDGES + e];
  int c0 = lane << 3;
  float acc[8];
#pragma unroll
  for (int j = 0; j < 8; ++j) acc[j] = 0.f;
#pragma unroll
  for (int t = 0; t < 16; ++t) {                 // e = edge_attr @ lin_edge_w
    float a = ea[(size_t)e * 16 + t];
    const float* w = &sWe[t * 256 + c0];
#pragma unroll
    for (int j = 0; j < 8; ++j) acc[j] += a * w[j];
  }
  const float4* pl = (const float4*)(xlr + (size_t)src * DHID + c0);        // x_l[src]
  const float4* pr = (const float4*)(xlr + (size_t)dst * DHID + 256 + c0);  // x_r[dst]
  float4 l0 = pl[0], l1 = pl[1], r0 = pr[0], r1 = pr[1];
  float mv[8] = { l0.x+r0.x+acc[0], l0.y+r0.y+acc[1], l0.z+r0.z+acc[2], l0.w+r0.w+acc[3],
                  l1.x+r1.x+acc[4], l1.y+r1.y+acc[5], l1.z+r1.z+acc[6], l1.w+r1.w+acc[7] };
  float p = 0.f;
#pragma unroll
  for (int j = 0; j < 8; ++j) {
    float m = mv[j];
    m = m > 0.f ? m : 0.2f * m;                  // leaky_relu(., 0.2)
    p += m * sAtt[c0 + j];
  }
  p += __shfl_xor(p, 1, 32);                     // reduce 4 lanes of one head
  p += __shfl_xor(p, 2, 32);
  if ((lane & 3) == 0) {
    int h = lane >> 2;
    logits[(size_t)e * 8 + h] = p;
    atomicMax(&segmax[(size_t)dst * 8 + h], f2ord(p));
  }
}

__global__ void k_edge_soft(const int* __restrict__ ei,
                            const float* __restrict__ logits,
                            const unsigned* __restrict__ segmax,
                            float* __restrict__ z, float* __restrict__ denom) {
  long t = (long)blockIdx.x * blockDim.x + threadIdx.x;
  if (t >= (long)N_EDGES * 8) return;
  int e = (int)(t >> 3), h = (int)(t & 7);
  int dst = ei[N_EDGES + e];
  float mx = ord2f(segmax[(size_t)dst * 8 + h]);
  float zz = __expf(logits[t] - mx);
  z[t] = zz;
  atomicAdd(&denom[(size_t)dst * 8 + h], zz);
}

__global__ void k_edge_aggr(const int* __restrict__ ei,
                            const float* __restrict__ xlr,
                            const float* __restrict__ z,
                            const float* __restrict__ denom,
                            float* __restrict__ outc) {
  int e = blockIdx.x * (blockDim.x >> 5) + (threadIdx.x >> 5);
  if (e >= N_EDGES) return;
  int lane = threadIdx.x & 31;
  int src = ei[e], dst = ei[N_EDGES + e];
  int c0 = lane << 3, h = lane >> 2;
  float alpha = z[(size_t)e * 8 + h] / denom[(size_t)dst * 8 + h];
  const float4* pl = (const float4*)(xlr + (size_t)src * DHID + c0);
  float4 l0 = pl[0], l1 = pl[1];
  float* o = outc + (size_t)dst * DIM + c0;
  atomicAdd(o + 0, l0.x * alpha); atomicAdd(o + 1, l0.y * alpha);
  atomicAdd(o + 2, l0.z * alpha); atomicAdd(o + 3, l0.w * alpha);
  atomicAdd(o + 4, l1.x * alpha); atomicAdd(o + 5, l1.y * alpha);
  atomicAdd(o + 6, l1.z * alpha); atomicAdd(o + 7, l1.w * alpha);
}

// ---------------------------------------------------------------- batch norm
__global__ void k_col_stats(const float* __restrict__ h, float* __restrict__ sum,
                            float* __restrict__ sq, int n, int cols) {
  int r0 = blockIdx.x * 128;
  int rend = r0 + 128 < n ? r0 + 128 : n;
  for (int c = threadIdx.x; c < cols; c += blockDim.x) {
    float s = 0.f, q = 0.f;
    for (int r = r0; r < rend; ++r) {
      float v = h[(size_t)r * cols + c];
      s += v; q += v * v;
    }
    atomicAdd(&sum[c], s);
    atomicAdd(&sq[c], q);
  }
}

__global__ void k_bn_apply(const float* __restrict__ h, const float* __restrict__ sum,
                           const float* __restrict__ sq, const float* __restrict__ g,
                           const float* __restrict__ b, bf16_t* __restrict__ dst,
                           int n, int cols) {
  long i = (long)blockIdx.x * blockDim.x + threadIdx.x;
  long tot = (long)n * cols;
  long stride = (long)gridDim.x * blockDim.x;
  float invN = 1.0f / (float)n;
  for (; i < tot; i += stride) {
    int c = (int)(i % cols);
    float mu = sum[c] * invN;
    float var = sq[c] * invN - mu * mu;
    float v = (h[i] - mu) * rsqrtf(var + EPSBN) * g[c] + b[c];
    dst[i] = f2bf(v);
  }
}

// ---------------------------------------------------------------------------
extern "C" void kernel_launch(void* const* d_in, const int* in_sizes, int n_in,
                              void* d_out, int out_size, void* d_ws, size_t ws_size,
                              hipStream_t stream) {
  const float* x       = (const float*)d_in[0];
  const int*   ei      = (const int*)  d_in[1];
  const float* ea      = (const float*)d_in[2];
  const float* lin_l_w = (const float*)d_in[3];
  const float* lin_l_b = (const float*)d_in[4];
  const float* lin_r_w = (const float*)d_in[5];
  const float* lin_r_b = (const float*)d_in[6];
  const float* lin_e_w = (const float*)d_in[7];
  const float* att     = (const float*)d_in[8];
  /* d_in[9] = conv_bias: cancels exactly inside BatchNorm1 (batch stats) */
  const float* bn1_g   = (const float*)d_in[10];
  const float* bn1_b   = (const float*)d_in[11];
  const float* w1      = (const float*)d_in[12];
  const float* b1      = (const float*)d_in[13];
  const float* bn2_g   = (const float*)d_in[14];
  const float* bn2_b   = (const float*)d_in[15];
  const float* w2      = (const float*)d_in[16];
  const float* b2      = (const float*)d_in[17];
  (void)in_sizes; (void)n_in; (void)out_size; (void)ws_size;

  char* ws = (char*)d_ws;
  size_t off = 0;
  auto carve = [&](size_t bytes) -> char* {
    char* p = ws + off;
    off = (off + bytes + 255) & ~(size_t)255;
    return p;
  };
  bf16_t*   xbf    = (bf16_t*) carve((size_t)N_NODES * DIM  * 2);
  bf16_t*   wlr_t  = (bf16_t*) carve((size_t)DHID * DIM * 2);   // [512][256]
  bf16_t*   w1_t   = (bf16_t*) carve((size_t)DHID * DIM * 2);   // [512][256]
  bf16_t*   w2_t   = (bf16_t*) carve((size_t)DIM * DHID * 2);   // [256][512]
  float*    xlr    = (float*)  carve((size_t)N_NODES * DHID * 4);  // [x_l|x_r]
  float*    logits = (float*)  carve((size_t)N_EDGES * 8 * 4);
  float*    zbuf   = (float*)  carve((size_t)N_EDGES * 8 * 4);
  unsigned* segmax = (unsigned*)carve((size_t)N_NODES * 8 * 4);
  float*    denom  = (float*)  carve((size_t)N_NODES * 8 * 4);
  float*    outc   = (float*)  carve((size_t)N_NODES * DIM * 4);
  float*    bn1s   = (float*)  carve(DIM * 4);
  float*    bn1q   = (float*)  carve(DIM * 4);
  bf16_t*   h1bf   = (bf16_t*) carve((size_t)N_NODES * DIM * 2);
  float*    gbuf   = (float*)  carve((size_t)N_NODES * DHID * 4);
  float*    bn2s   = (float*)  carve(DHID * 4);
  float*    bn2q   = (float*)  carve(DHID * 4);
  bf16_t*   h2bf   = (bf16_t*) carve((size_t)N_NODES * DHID * 2);

  // zero accumulators every call (graph-capturable async memsets)
  hipMemsetAsync(segmax, 0, (size_t)N_NODES * 8 * 4, stream);
  hipMemsetAsync(denom,  0, (size_t)N_NODES * 8 * 4, stream);
  hipMemsetAsync(outc,   0, (size_t)N_NODES * DIM * 4, stream);
  hipMemsetAsync(bn1s,   0, DIM * 4, stream);
  hipMemsetAsync(bn1q,   0, DIM * 4, stream);
  hipMemsetAsync(bn2s,   0, DHID * 4, stream);
  hipMemsetAsync(bn2q,   0, DHID * 4, stream);

  // precision conversions / weight transposes
  k_f32_to_bf16<<<2048, 256, 0, stream>>>(x, xbf, (long)N_NODES * DIM);
  k_transpose_bf16<<<(256 * 256) / 256, 256, 0, stream>>>(lin_l_w, wlr_t, 256, 256);
  k_transpose_bf16<<<(256 * 256) / 256, 256, 0, stream>>>(lin_r_w, wlr_t + 256 * 256, 256, 256);
  k_transpose_bf16<<<(256 * 512) / 256, 256, 0, stream>>>(w1, w1_t, 256, 512);
  k_transpose_bf16<<<(512 * 256) / 256, 256, 0, stream>>>(w2, w2_t, 512, 256);

  // GEMM1: xlr[N,512] = x @ [lin_l_w | lin_r_w] + [lin_l_b | lin_r_b]
  {
    int strips = (N_NODES / 16) * (DHID / 64);
    k_wmma_gemm<<<(strips + 7) / 8, 256, 0, stream>>>(
        xbf, wlr_t, xlr, N_NODES, DHID, DIM, 0, lin_l_b, lin_r_b, nullptr);
  }

  // edge phase: logits + segment max -> softmax denom -> scatter aggregate
  k_edge_logits<<<N_EDGES / 8, 256, 0, stream>>>(ei, ea, xlr, lin_e_w, att,
                                                 logits, segmax);
  k_edge_soft<<<(N_EDGES * 8) / 256, 256, 0, stream>>>(ei, logits, segmax,
                                                       zbuf, denom);
  k_edge_aggr<<<N_EDGES / 8, 256, 0, stream>>>(ei, xlr, zbuf, denom, outc);

  // MLP: BN1 -> Linear(256,512)+LReLU -> BN2 -> Linear(512,256)+LReLU -> +x
  k_col_stats<<<(N_NODES + 127) / 128, 256, 0, stream>>>(outc, bn1s, bn1q,
                                                         N_NODES, DIM);
  k_bn_apply<<<4096, 256, 0, stream>>>(outc, bn1s, bn1q, bn1_g, bn1_b, h1bf,
                                       N_NODES, DIM);
  {
    int strips = (N_NODES / 16) * (DHID / 64);
    k_wmma_gemm<<<(strips + 7) / 8, 256, 0, stream>>>(
        h1bf, w1_t, gbuf, N_NODES, DHID, DIM, 1, b1, nullptr, nullptr);
  }
  k_col_stats<<<(N_NODES + 127) / 128, 256, 0, stream>>>(gbuf, bn2s, bn2q,
                                                         N_NODES, DHID);
  k_bn_apply<<<4096, 256, 0, stream>>>(gbuf, bn2s, bn2q, bn2_g, bn2_b, h2bf,
                                       N_NODES, DHID);
  {
    int strips = (N_NODES / 16) * (DIM / 64);
    k_wmma_gemm<<<(strips + 7) / 8, 256, 0, stream>>>(
        h2bf, w2_t, (float*)d_out, N_NODES, DIM, DHID, 2, b2, nullptr, x);
  }
}
